// GraphAttentionLayerV2_25520695673170
// MI455X (gfx1250) — matrix operand order repaired and checked by reading, not compiled, not run on hardware
//
#include <hip/hip_runtime.h>
#include <hip/hip_bf16.h>

// GATv2 layer, MI455X (gfx1250, wave32, WMMA).
// N=2048, F_IN=512, F_OUT=64.
#define NV    2048
#define FIN   512
#define FOUT  64
#define ALPHA 0.2f
#define NEGI  (-9e15f)

#define ATTP  2052   // att row stride (floats): bank=(4*row+k)%64, 16B-aligned
#define WPAD  68     // staging row stride (floats): bank=(4*row+f)%64, 16B-aligned

typedef __attribute__((ext_vector_type(16))) __bf16 v16bf;
typedef __attribute__((ext_vector_type(8)))  __bf16 v8bf;
typedef __attribute__((ext_vector_type(8)))  float  v8f;

__device__ __forceinline__ v16bf cat8(v8bf lo, v8bf hi) {
  return __builtin_shufflevector(lo, hi, 0,1,2,3,4,5,6,7,8,9,10,11,12,13,14,15);
}

// ---------------------------------------------------------------------------
// Kernel 1: h -> bf16 row-major [2048][512];
//           Wt bf16 [128][512]: Wt[c][k] = (c<64 ? W[k][c] : W[512+k][c-64])
// ---------------------------------------------------------------------------
__global__ __launch_bounds__(256) void k_convert(
    const float* __restrict__ h, const float* __restrict__ W,
    __bf16* __restrict__ hb, __bf16* __restrict__ Wt) {
  int tid = blockIdx.x * blockDim.x + threadIdx.x;
  int stride = gridDim.x * blockDim.x;
  for (int idx = tid; idx < NV * FIN; idx += stride)
    hb[idx] = (__bf16)h[idx];
  for (int idx = tid; idx < 2 * FOUT * FIN; idx += stride) {
    int c = idx / FIN, k = idx % FIN;
    float v = (c < FOUT) ? W[k * FOUT + c] : W[(FIN + k) * FOUT + (c - FOUT)];
    Wt[idx] = (__bf16)v;
  }
}

// ---------------------------------------------------------------------------
// Kernel 2: projection GEMM via v_wmma_f32_16x16x32_bf16.
//   Whc f32 [2048][128] = h @ [W1 | W2]  (K = 512)
//   Also emit Wh2t bf16 [64][2048] (transposed key/value projection).
// ---------------------------------------------------------------------------
__global__ __launch_bounds__(256) void k_proj(
    const __bf16* __restrict__ hb, const __bf16* __restrict__ Wt,
    float* __restrict__ Whc, __bf16* __restrict__ Wh2t) {
  int m0   = blockIdx.x * 16;
  int lane = threadIdx.x & 31;
  int wave = threadIdx.x >> 5;   // 0..7 -> output columns [wave*16, wave*16+16)
  int hh   = lane >> 4;
  int lrow = lane & 15;
  int c    = wave * 16 + lrow;   // 0..127

  v8f acc = {};
  for (int k0 = 0; k0 < FIN; k0 += 32) {
    // A 16x32 bf16: lane<16 holds K {0..7,16..23}; lane>=16 holds {8..15,24..31}
    const __bf16* pa = hb + (m0 + lrow) * FIN + k0 + hh * 8;
    v16bf A = cat8(*(const v8bf*)pa, *(const v8bf*)(pa + 16));
    // B 32x16 bf16 (transposed operand): lane<16 -> K {0..7,16..23}, lane>=16 -> {8..15,24..31}
    const __bf16* pb = Wt + c * FIN + k0 + hh * 8;
    v16bf B = cat8(*(const v8bf*)pb, *(const v8bf*)(pb + 16));
    acc = __builtin_amdgcn_wmma_f32_16x16x32_bf16(
        false, A, false, B, (short)0, acc, false, false);
  }
#pragma unroll
  for (int v = 0; v < 8; ++v) {
    int m = m0 + v + hh * 8;            // C: VGPR v -> M = v + 8*hh, N = lrow
    float val = acc[v];
    Whc[m * 128 + c] = val;
    if (c >= FOUT) Wh2t[(c - FOUT) * NV + m] = (__bf16)val;
  }
}

// ---------------------------------------------------------------------------
// Kernel 3: fused scores + softmax + aggregation + ELU, one 16-row block/WG.
// LDS ~153 KB (att padded to stride 2052 for conflict-free b128 reads).
// ---------------------------------------------------------------------------
#define TK 64
__global__ __launch_bounds__(256) void k_attn(
    const float* __restrict__ Whc, const int* __restrict__ adj,
    const float* __restrict__ avec, const __bf16* __restrict__ Wh2t,
    float* __restrict__ out) {
  __shared__ __align__(16) float att[16][ATTP];     // ~128 KB
  __shared__ __align__(16) float wh1s[16][WPAD];
  __shared__ __align__(16) float wh2s[TK][WPAD];
  __shared__ float as_[FOUT];

  int tid = threadIdx.x;
  int i0  = blockIdx.x * 16;

  // wh1 block: 16 rows x 16 float4 -> one float4 per thread
  {
    int i = tid >> 4, f4 = tid & 15;
    float4 v = *(const float4*)(Whc + (i0 + i) * 128 + f4 * 4);
    *(float4*)&wh1s[i][f4 * 4] = v;
  }
  if (tid < FOUT) as_[tid] = avec[tid];
  __syncthreads();

  int i  = tid >> 4;                 // 0..15  (query row within block)
  int kl = tid & 15;
  float wh1r[FOUT];
#pragma unroll
  for (int f4 = 0; f4 < FOUT / 4; ++f4) {
    float4 v = *(const float4*)&wh1s[i][f4 * 4];
    wh1r[f4 * 4 + 0] = v.x; wh1r[f4 * 4 + 1] = v.y;
    wh1r[f4 * 4 + 2] = v.z; wh1r[f4 * 4 + 3] = v.w;
  }
  const int* adjrow = adj + (i0 + i) * NV;

  // -------- Phase A: e[i,k] for all k, masked --------
  for (int k0 = 0; k0 < NV; k0 += TK) {
    __syncthreads();
    // cooperative tile load: 64 rows x 16 float4 = 1024 float4 / 256 thr
    for (int idx = tid; idx < TK * 16; idx += 256) {
      int kk = idx >> 4, f4 = idx & 15;
      float4 v = *(const float4*)(Whc + (k0 + kk) * 128 + FOUT + f4 * 4);
      *(float4*)&wh2s[kk][f4 * 4] = v;
    }
    if (k0 + TK < NV)  // global_prefetch_b8 hint for next tile
      __builtin_prefetch(Whc + (k0 + TK) * 128 + FOUT, 0, 3);
    __syncthreads();
#pragma unroll
    for (int j = 0; j < TK / 16; ++j) {
      int kk = kl + j * 16;
      int k  = k0 + kk;
      float acc = 0.f;
#pragma unroll
      for (int f4 = 0; f4 < FOUT / 4; ++f4) {
        float4 w2 = *(const float4*)&wh2s[kk][f4 * 4];   // ds_load_b128
        float s0 = wh1r[f4 * 4 + 0] + w2.x;
        float s1 = wh1r[f4 * 4 + 1] + w2.y;
        float s2 = wh1r[f4 * 4 + 2] + w2.z;
        float s3 = wh1r[f4 * 4 + 3] + w2.w;
        acc = fmaf(fmaxf(s0, ALPHA * s0), as_[f4 * 4 + 0], acc);
        acc = fmaf(fmaxf(s1, ALPHA * s1), as_[f4 * 4 + 1], acc);
        acc = fmaf(fmaxf(s2, ALPHA * s2), as_[f4 * 4 + 2], acc);
        acc = fmaf(fmaxf(s3, ALPHA * s3), as_[f4 * 4 + 3], acc);
      }
      if (adjrow[k] <= 0) acc = NEGI;
      att[i][k] = acc;
    }
  }
  __syncthreads();

  // -------- Phase B: softmax per row (wave handles 2 rows, wave32 shuffles) --------
  int wave = tid >> 5, lane = tid & 31;
  for (int r = wave * 2; r < wave * 2 + 2; ++r) {
    float m = -3.4e38f;
    for (int x = lane; x < NV; x += 32) m = fmaxf(m, att[r][x]);
#pragma unroll
    for (int off = 16; off > 0; off >>= 1) m = fmaxf(m, __shfl_xor(m, off, 32));
    float ssum = 0.f;
    for (int x = lane; x < NV; x += 32) {
      float e = __expf(att[r][x] - m);
      att[r][x] = e;
      ssum += e;
    }
#pragma unroll
    for (int off = 16; off > 0; off >>= 1) ssum += __shfl_xor(ssum, off, 32);
    float inv = 1.f / ssum;
    for (int x = lane; x < NV; x += 32) att[r][x] *= inv;
  }
  __syncthreads();

  // -------- Phase C: h' = att @ Wh2 via WMMA bf16, then ELU --------
  if (wave < 4) {                       // wave-uniform: EXEC all-ones in WMMA
    int hh   = lane >> 4;
    int lrow = lane & 15;
    int n    = wave * 16 + lrow;        // output feature column
    v8f c = {};
    for (int k0 = 0; k0 < NV; k0 += 32) {
      int kb = k0 + hh * 8;
      // A elems 0..7 = att[lrow][kb..kb+7], 8..15 = att[lrow][kb+16..kb+23]
      float4 a0 = *(const float4*)&att[lrow][kb];
      float4 a1 = *(const float4*)&att[lrow][kb + 4];
      float4 a2 = *(const float4*)&att[lrow][kb + 16];
      float4 a3 = *(const float4*)&att[lrow][kb + 20];
      v16bf A;
      A[0]  = (__bf16)a0.x; A[1]  = (__bf16)a0.y; A[2]  = (__bf16)a0.z; A[3]  = (__bf16)a0.w;
      A[4]  = (__bf16)a1.x; A[5]  = (__bf16)a1.y; A[6]  = (__bf16)a1.z; A[7]  = (__bf16)a1.w;
      A[8]  = (__bf16)a2.x; A[9]  = (__bf16)a2.y; A[10] = (__bf16)a2.z; A[11] = (__bf16)a2.w;
      A[12] = (__bf16)a3.x; A[13] = (__bf16)a3.y; A[14] = (__bf16)a3.z; A[15] = (__bf16)a3.w;
      const __bf16* pb = Wh2t + n * NV + k0 + hh * 8;
      v16bf B = cat8(*(const v8bf*)pb, *(const v8bf*)(pb + 16));
      c = __builtin_amdgcn_wmma_f32_16x16x32_bf16(
          false, A, false, B, (short)0, c, false, false);
    }
#pragma unroll
    for (int v = 0; v < 8; ++v) {
      int m = v + hh * 8;
      float x = c[v];
      float y = (x > 0.f) ? x : (__expf(x) - 1.f);   // ELU(alpha=1)
      out[(i0 + m) * FOUT + n] = y;
    }
  }
}

// ---------------------------------------------------------------------------
extern "C" void kernel_launch(void* const* d_in, const int* in_sizes, int n_in,
                              void* d_out, int out_size, void* d_ws, size_t ws_size,
                              hipStream_t stream) {
  const float* h   = (const float*)d_in[0];   // [2048][512]
  const int*   adj = (const int*)  d_in[1];   // [2048][2048]
  const float* W   = (const float*)d_in[2];   // [1024][64]
  const float* a   = (const float*)d_in[3];   // [64][1]
  float* out = (float*)d_out;                 // [2048][64]

  char* ws = (char*)d_ws;
  __bf16* hb   = (__bf16*)ws;                                   // 2 MB
  __bf16* Wt   = (__bf16*)(ws + (2u << 20));                    // 128 KB
  float*  Whc  = (float*) (ws + (2u << 20) + (128u << 10));     // 1 MB
  __bf16* Wh2t = (__bf16*)(ws + (3u << 20) + (128u << 10));     // 256 KB

  k_convert<<<512, 256, 0, stream>>>(h, W, hb, Wt);
  k_proj   <<<NV / 16, 256, 0, stream>>>(hb, Wt, Whc, Wh2t);
  k_attn   <<<NV / 16, 256, 0, stream>>>(Whc, adj, a, Wh2t, out);
}
